// VectorQuantizer3D_69647189671950
// MI455X (gfx1250) — matrix operand order repaired and checked by reading, not compiled, not run on hardware
//
#include <hip/hip_runtime.h>
#include <stdint.h>

// ---------------- problem constants ----------------
#define N_TOT 16384      // 2*8*32*32 flattened vectors
#define EDIM  256
#define NE    4096

// d_out flat layout (all float32), in reference return order
static constexpr size_t O_LOSS   = 0;
static constexpr size_t O_ZQ     = 1;                    // 4,194,304
static constexpr size_t O_PERP   = 4194305ull;
static constexpr size_t O_ONEHOT = 4194306ull;           // 67,108,864
static constexpr size_t O_IDX    = 71303170ull;          // 16,384
static constexpr size_t O_ZOUT   = 71319554ull;          // 4,194,304
static constexpr size_t O_EMA    = 75513858ull;          // 1,048,576

// ws byte offsets
static constexpr size_t WS_ZBF   = 0;          // ushort[16384*256]  = 8 MB
static constexpr size_t WS_EBF   = 8388608;    // ushort[4096*256]   = 2 MB
static constexpr size_t WS_ENORM = 10485760;   // float[4096]
static constexpr size_t WS_IDX   = 10502144;   // int[16384]
static constexpr size_t WS_CNT   = 10567680;   // int[4096]
static constexpr size_t WS_LP    = 10584064;   // float[512]

typedef __attribute__((ext_vector_type(16))) __bf16 v16bf;
typedef __attribute__((ext_vector_type(8)))  float  v8f;

__device__ __forceinline__ unsigned short f2bf(float f) {
  union { float f; uint32_t u; } x; x.f = f;
  uint32_t u = x.u;
  u += 0x7FFFu + ((u >> 16) & 1u);   // round-to-nearest-even
  return (unsigned short)(u >> 16);
}

// ---------------- K0: zero one-hot region (268 MB) + histogram ----------------
__global__ void __launch_bounds__(256) k_zero(float2* __restrict__ onehot,
                                              int* __restrict__ counts) {
  size_t i = (size_t)blockIdx.x * 256 + threadIdx.x;   // 33,554,432 float2
  float2 z2; z2.x = 0.0f; z2.y = 0.0f;
  onehot[i] = z2;
  if (i < NE) counts[i] = 0;
}

// ---------------- K1: codebook -> bf16, fused EMA output ----------------
__global__ void __launch_bounds__(256) k_emb(const float* __restrict__ emb,
                                             const float* __restrict__ ema,
                                             unsigned short* __restrict__ ebf,
                                             float* __restrict__ emaOut) {
  int i = blockIdx.x * 256 + threadIdx.x;              // NE*EDIM = 1,048,576
  float e = emb[i];
  ebf[i] = f2bf(e);
  emaOut[i] = 0.25f * ema[i] + 0.75f * e;              // BETA*ema + (1-BETA)*W
}

// ---------------- K1b: exact f32 codebook norms ----------------
__global__ void __launch_bounds__(256) k_enorm(const float* __restrict__ emb,
                                               float* __restrict__ enorm) {
  int wave = (blockIdx.x * 256 + threadIdx.x) >> 5;    // one wave32 per row
  int lane = threadIdx.x & 31;
  const float* row = emb + (size_t)wave * EDIM;
  float s = 0.0f;
  #pragma unroll
  for (int k = 0; k < EDIM / 32; ++k) { float v = row[lane + k * 32]; s += v * v; }
  #pragma unroll
  for (int off = 16; off >= 1; off >>= 1) s += __shfl_xor(s, off, 32);
  if (lane == 0) enorm[wave] = s;
}

// ---------------- K_tr: z[b][c][m] -> zbf[b*8192+m][c] (bf16), tiled ----------------
__global__ void __launch_bounds__(256) k_ztr(const float* __restrict__ z,
                                             unsigned short* __restrict__ zbf) {
  __shared__ float t[32][33];
  int b  = blockIdx.z;
  int m0 = blockIdx.x * 32;
  int c0 = blockIdx.y * 32;
  int tx = threadIdx.x, ty = threadIdx.y;              // block (32, 8)
  #pragma unroll
  for (int j = 0; j < 4; ++j) {
    int c = c0 + ty + j * 8;
    t[ty + j * 8][tx] = z[((size_t)(b * 256 + c)) * 8192 + m0 + tx]; // coalesced read
  }
  __syncthreads();
  #pragma unroll
  for (int j = 0; j < 4; ++j) {
    int m = m0 + ty + j * 8;
    zbf[((size_t)(b * 8192 + m)) * EDIM + c0 + tx] = f2bf(t[tx][ty + j * 8]); // coalesced write
  }
}

// ---------------- K2: fused GEMM + argmin ----------------
// One wave32 handles 16 rows of zf against all 4096 codes.
// Codebook chunks (16 codes x 256 K, bf16) are staged into LDS ONCE per block
// (8 waves share them) with double-buffered async copies (ASYNCcnt), and the
// 8 v_wmma_f32_16x16x32_bf16 per chunk read B fragments from LDS.
// argmin_n( ||e_n||^2 - 2 * z.e_n )   (||z||^2 dropped: row-constant)
#define BROW 264   // 256 data bf16 + 8 pad -> 528B row stride = 132 dwords (bank stride 4)

__global__ void __launch_bounds__(256) k_argmin(const unsigned short* __restrict__ zbf,
                                                const unsigned short* __restrict__ ebf,
                                                const float* __restrict__ enorm,
                                                int* __restrict__ idxOut) {
  __shared__ unsigned short bsm[2][16][BROW];          // 2 x 8.25 KB double buffer
  int tid   = threadIdx.x;
  int lane  = tid & 31;
  int gwave = blockIdx.x * 8 + (tid >> 5);
  int row0  = gwave * 16;

  // --- preload 8 A-fragments (K = 0..255), invariant over the code loop ---
  // A layout used here (contiguous k; B uses the same): lane m = lane&15; lanes 0-15 hold K 0..15,
  // lanes 16-31 hold K 16..31 (per 32-wide K chunk).
  int m    = row0 + (lane & 15);
  int koff = (lane >> 4) * 16;
  const unsigned short* zrow = zbf + (size_t)m * EDIM;
  union AU { v16bf v; uint4 q[2]; };
  v16bf afr[8];
  #pragma unroll
  for (int kk = 0; kk < 8; ++kk) {
    AU u;
    u.q[0] = *(const uint4*)(zrow + kk * 32 + koff);        // K = 32kk+koff .. +7
    u.q[1] = *(const uint4*)(zrow + kk * 32 + koff + 8);    // K = 32kk+koff+8 .. +15
    afr[kk] = u.v;
  }

  // stage one 16-code chunk into LDS buffer `buf` with async b128 copies:
  // 512 x 16B pieces, 2 per thread; both sides 16B-aligned.
  auto stage = [&](int buf, int nb) {
    #pragma unroll
    for (int h = 0; h < 2; ++h) {
      int c    = tid + h * 256;
      int row  = c >> 5;
      int p16  = (c & 31) * 8;                              // piece offset in ushorts
      unsigned ldsAddr = (unsigned)(uintptr_t)&bsm[buf][row][p16];
      const unsigned short* g = ebf + ((size_t)(nb + row)) * EDIM + p16;
      asm volatile("global_load_async_to_lds_b128 %0, %1, off"
                   :: "v"(ldsAddr), "v"(g) : "memory");
    }
  };

  // B layout used here (32x16, KxN, same k order as A): lane n = lane&15; lanes 0-15 hold K 0..15,
  // lanes 16-31 hold K 16..31 -> 16 contiguous bf16 per lane (2x ds_load_b128).
  int nlane = lane & 15;
  int khalf = lane >> 4;

  float minv[8]; int mini[8];
  #pragma unroll
  for (int r = 0; r < 8; ++r) { minv[r] = 3.0e38f; mini[r] = 0; }

  stage(0, 0);
  asm volatile("s_wait_asynccnt 0x0" ::: "memory");
  __syncthreads();

  for (int nb = 0; nb < NE; nb += 16) {
    int cur = (nb >> 4) & 1;
    if (nb + 16 < NE) stage(cur ^ 1, nb + 16);              // overlap copy with compute
    const unsigned short* brow = &bsm[cur][nlane][khalf * 16];
    v8f acc = {};
    #pragma unroll
    for (int kk = 0; kk < 8; ++kk) {
      v16bf bfr = *(const v16bf*)(brow + kk * 32);
      acc = __builtin_amdgcn_wmma_f32_16x16x32_bf16(
          /*neg_a=*/false, afr[kk], /*neg_b=*/false, bfr,
          /*c_mod=*/(short)0, acc, /*reuse_a=*/false, /*reuse_b=*/false);
    }
    float en  = enorm[nb + nlane];
    int  code = nb + nlane;
    #pragma unroll
    for (int r = 0; r < 8; ++r) {
      float d = en - 2.0f * acc[r];
      if (d < minv[r]) { minv[r] = d; mini[r] = code; }     // first-index on ties
    }
    if (nb + 16 < NE)
      asm volatile("s_wait_asynccnt 0x0" ::: "memory");     // my copies landed
    __syncthreads();                                        // everyone done (reads + copies)
  }

  // C layout: VGPR r = row (row0+r) on lanes 0-15, row (row0+8+r) on lanes 16-31.
  // Reduce across the 16 lanes of each half (xor masks 8..1 stay in-half).
  #pragma unroll
  for (int r = 0; r < 8; ++r) {
    #pragma unroll
    for (int off = 8; off >= 1; off >>= 1) {
      float ov = __shfl_xor(minv[r], off, 32);
      int   oi = __shfl_xor(mini[r], off, 32);
      if (ov < minv[r] || (ov == minv[r] && oi < mini[r])) { minv[r] = ov; mini[r] = oi; }
    }
  }
  if ((lane & 15) == 0) {
    int base = row0 + khalf * 8;
    #pragma unroll
    for (int r = 0; r < 8; ++r) idxOut[base + r] = mini[r];
  }
}

// ---------------- K3: indices (as float), one-hot scatter, histogram ----------------
__global__ void __launch_bounds__(256) k_scatter(const int* __restrict__ idx,
                                                 float* __restrict__ out,
                                                 int* __restrict__ counts) {
  int n  = blockIdx.x * 256 + threadIdx.x;
  int id = idx[n];
  out[O_IDX + n] = (float)id;
  out[O_ONEHOT + (size_t)n * NE + id] = 1.0f;
  atomicAdd(&counts[id], 1);
}

// ---------------- K4: gather z_q, write z_q_st & z_out, exact f32 loss partials ----------------
__global__ void __launch_bounds__(256) k_gather(const float* __restrict__ z,
                                                const float* __restrict__ emb,
                                                const int* __restrict__ idx,
                                                float* __restrict__ out,
                                                float* __restrict__ lossPart) {
  __shared__ float tile[32][257];
  __shared__ float red[256];
  int b   = blockIdx.y;
  int m0  = blockIdx.x * 32;
  int tid = threadIdx.x;

  for (int r = 0; r < 32; ++r) {
    int rowIdx = idx[b * 8192 + m0 + r];                       // broadcast
    tile[r][tid] = emb[(size_t)rowIdx * EDIM + tid];           // coalesced gather
  }
  __syncthreads();

  int mloc  = tid & 31;
  int cbase = tid >> 5;
  float s = 0.0f;
  for (int j = 0; j < 32; ++j) {
    int c = cbase * 32 + j;
    size_t o = ((size_t)(b * 256 + c)) * 8192 + m0 + mloc;     // [B,C,D,H,W] flat
    float e  = tile[mloc][c];                                  // conflict-free (stride 257)
    float zv = z[o];
    float d  = e - zv;                                         // z_q - zp
    s += d * d;
    out[O_ZQ + o]   = zv + d;                                  // straight-through: zp + (z_q - zp)
    out[O_ZOUT + o] = zv;                                      // z_out == z
  }
  red[tid] = s;
  __syncthreads();
  for (int st = 128; st >= 1; st >>= 1) { if (tid < st) red[tid] += red[tid + st]; __syncthreads(); }
  if (tid == 0) lossPart[blockIdx.y * gridDim.x + blockIdx.x] = red[0];  // fixed slot -> deterministic
}

// ---------------- K5: finalize loss + perplexity ----------------
__global__ void __launch_bounds__(256) k_final(const float* __restrict__ lossPart,
                                               const int* __restrict__ counts,
                                               float* __restrict__ out) {
  __shared__ float red[256];
  int tid = threadIdx.x;
  red[tid] = lossPart[tid] + lossPart[tid + 256];
  __syncthreads();
  for (int st = 128; st >= 1; st >>= 1) { if (tid < st) red[tid] += red[tid + st]; __syncthreads(); }
  float lossSum = red[0];
  __syncthreads();
  float h = 0.0f;
  #pragma unroll
  for (int k = 0; k < NE / 256; ++k) {
    float p = (float)counts[tid + k * 256] * (1.0f / (float)N_TOT);
    h += p * logf(p + 1e-10f);
  }
  red[tid] = h;
  __syncthreads();
  for (int st = 128; st >= 1; st >>= 1) { if (tid < st) red[tid] += red[tid + st]; __syncthreads(); }
  if (tid == 0) {
    out[O_LOSS] = 1.25f * lossSum * (1.0f / ((float)N_TOT * (float)EDIM));  // (1+BETA)*mse
    out[O_PERP] = expf(-red[0]);
  }
}

// ---------------- launcher ----------------
extern "C" void kernel_launch(void* const* d_in, const int* in_sizes, int n_in,
                              void* d_out, int out_size, void* d_ws, size_t ws_size,
                              hipStream_t stream) {
  const float* z   = (const float*)d_in[0];
  const float* emb = (const float*)d_in[1];
  const float* ema = (const float*)d_in[2];
  float* out = (float*)d_out;
  char*  ws  = (char*)d_ws;

  unsigned short* zbf   = (unsigned short*)(ws + WS_ZBF);
  unsigned short* ebf   = (unsigned short*)(ws + WS_EBF);
  float*          enorm = (float*)(ws + WS_ENORM);
  int*            idx   = (int*)(ws + WS_IDX);
  int*            cnt   = (int*)(ws + WS_CNT);
  float*          lp    = (float*)(ws + WS_LP);

  // 268 MB one-hot zero-fill (float2: region byte offset is 8-aligned, not 16)
  k_zero<<<131072, 256, 0, stream>>>((float2*)(out + O_ONEHOT), cnt);
  k_emb<<<4096, 256, 0, stream>>>(emb, ema, ebf, out + O_EMA);
  k_enorm<<<512, 256, 0, stream>>>(emb, enorm);
  k_ztr<<<dim3(256, 8, 2), dim3(32, 8), 0, stream>>>(z, zbf);
  k_argmin<<<128, 256, 0, stream>>>(zbf, ebf, enorm, idx);
  k_scatter<<<64, 256, 0, stream>>>(idx, out, cnt);
  k_gather<<<dim3(256, 2), 256, 0, stream>>>(z, emb, idx, out, lp);
  k_final<<<1, 256, 0, stream>>>(lp, cnt, out);
}
